// RNNModel_12979391168914
// MI455X (gfx1250) — matrix operand order repaired and checked by reading, not compiled, not run on hardware
//
#include <hip/hip_runtime.h>

// ---------------------------------------------------------------------------
// Fused hardtanh-RNN for MI455X (gfx1250, wave32).
//
//   pre_t = xs_t @ W_in^T          (fused, never materialized: saves ~1 GB HBM)
//   h_t   = clip(pre_t + h_{t-1} @ W_h^T, -1, 1)   (T=2048 sequential steps)
//   out   = h_T @ W_out^T
//
// 64 workgroups x (16 batch rows); 8 waves/WG, each wave owns a 16-wide
// N-slice of H=128.  Hidden state (16x128 bf16) double-buffered in LDS.
// Per step per wave: 1 bf16 WMMA (input proj, D=13 padded into one K=32 tile)
// + 4 bf16 WMMAs (K=128 recurrence), f32 accumulators, hardtanh, bf16
// writeback.  W_in/W_h B-fragments live in registers for the whole scan.
// xs loads are software-pipelined one step ahead; all LDS A-fragments are
// preloaded before the WMMA chain so dscnt waits can be partial.
// ---------------------------------------------------------------------------

typedef __attribute__((ext_vector_type(16))) __bf16 v16bf;
typedef __attribute__((ext_vector_type(8)))  __bf16 v8bf;
typedef __attribute__((ext_vector_type(8)))  float  v8f;

// 16-byte chunk with natural 4-byte alignment (xs row stride = 13 floats,
// so 16B alignment cannot be assumed and we must not overrun row end).
struct f4 { float v[4]; };

#define B_SZ 1024
#define T_SZ 2048
#define D_SZ 13
#define H_SZ 128
#define O_SZ 2

__global__ __launch_bounds__(256) void rnn_hardtanh_fused(
    const float* __restrict__ xs,     // [B, T, D]
    const float* __restrict__ W_in,   // [H, D]
    const float* __restrict__ W_h,    // [H, H]
    const float* __restrict__ W_out,  // [O, H]
    float* __restrict__ out)          // [B, O]
{
    __shared__ __align__(16) __bf16 hbuf[2][16][H_SZ];   // 8 KB, double-buffered h
    __shared__ __align__(16) float  hf[16][H_SZ];        // 8 KB, final f32 h

    const int wave = threadIdx.x >> 5;          // 0..7 : N-tile of hidden dim
    const int lane = threadIdx.x & 31;
    const int lrow = lane & 15;                 // M row (A) or N column (B/C)
    const bool hi  = lane >= 16;
    const int  b0  = blockIdx.x * 16;           // batch tile base
    const int  n0  = wave * 16;                 // hidden-column tile base

    // ---- Preload B-operand fragments (registers, loop-invariant) ------------
    // B layout (32x16 bf16): b_win n<16 K=0..7,16..23, n>=16 K=8..15,24..31; b_wh n<16 K=0..15, n>=16 K=16..31,
    // 2 bf16 per VGPR.  B[k][n] = W[n0+n][k], b_wh contiguous per lane.
    v16bf b_win;
    {
        const int n = n0 + lrow;
        const int kbase = hi ? 8 : 0;
#pragma unroll
        for (int j = 0; j < 16; ++j) {
            const int k = kbase + (j < 8 ? j : j + 8);
            b_win[j] = (k < D_SZ) ? (__bf16)W_in[n * D_SZ + k] : (__bf16)0.0f;
        }
    }
    v16bf b_wh[4];
#pragma unroll
    for (int kk = 0; kk < 4; ++kk) {
        const int n = n0 + lrow;
        const int kbase = kk * 32 + (hi ? 16 : 0);
#pragma unroll
        for (int j = 0; j < 16; ++j)
            b_wh[kk][j] = (__bf16)W_h[n * H_SZ + kbase + j];
    }

    // ---- h0 = 0 --------------------------------------------------------------
    for (int i = threadIdx.x; i < 2 * 16 * H_SZ; i += 256)
        ((__bf16*)hbuf)[i] = (__bf16)0.0f;
    __syncthreads();

    // ---- xs pipeline: lane's row base; lo lanes carry k=0..7, hi k=8..12 ----
    const float* xrow0 = xs + (size_t)(b0 + lrow) * T_SZ * D_SZ;

    f4 p0 = {}, p1 = {};
    float p2 = 0.0f;
    {   // prologue load for t = 0
        const float* r = xrow0;
        if (!hi) { p0 = *(const f4*)r;       p1 = *(const f4*)(r + 4); }
        else     { p0 = *(const f4*)(r + 8); p2 = r[12]; }
    }

    v8f c = {};   // final clamped h tile (f32) survives the loop

    for (int t = 0; t < T_SZ; ++t) {
        const int cur = t & 1;
        const int prv = cur ^ 1;

        // -- A_x from pipelined registers (K padded 13 -> 32) -----------------
        // A layout: lane<16 -> m=lane,    chunks K=0..7  and K=16..23
        //           lane>=16 -> m=lane-16, chunks K=8..15 and K=24..31
        v16bf ax;
#pragma unroll
        for (int j = 0; j < 4; ++j) ax[j] = (__bf16)p0.v[j];
        ax[4] = (__bf16)(hi ? p2 : p1.v[0]);       // k=12 for hi lanes
        ax[5] = hi ? (__bf16)0.0f : (__bf16)p1.v[1];
        ax[6] = hi ? (__bf16)0.0f : (__bf16)p1.v[2];
        ax[7] = hi ? (__bf16)0.0f : (__bf16)p1.v[3];
#pragma unroll
        for (int j = 8; j < 16; ++j) ax[j] = (__bf16)0.0f;   // K=16..31 padding

        // -- issue NEXT step's xs loads now; they hide behind WMMAs + barrier -
        f4 q0 = {}, q1 = {};
        float q2 = 0.0f;
        if (t + 1 < T_SZ) {
            const float* r = xrow0 + (size_t)(t + 1) * D_SZ;
            if (!hi) { q0 = *(const f4*)r;       q1 = *(const f4*)(r + 4); }
            else     { q0 = *(const f4*)(r + 8); q2 = r[12]; }
            if (t + 2 < T_SZ)   // keep t+2's line warm (global_prefetch_b8)
                __builtin_prefetch(xrow0 + (size_t)(t + 2) * D_SZ, 0, 0);
        }

        // -- preload ALL recurrence A-fragments, then chain the 5 WMMAs -------
        v16bf ah[4];
#pragma unroll
        for (int kk = 0; kk < 4; ++kk) {
            const int klo = kk * 32 + (hi ? 16 : 0);
            const v8bf lo8 = *(const v8bf*)&hbuf[prv][lrow][klo];
            const v8bf hi8 = *(const v8bf*)&hbuf[prv][lrow][klo + 8];
#pragma unroll
            for (int j = 0; j < 8; ++j) { ah[kk][j] = lo8[j]; ah[kk][8 + j] = hi8[j]; }
        }

        v8f cz = {};
        c = __builtin_amdgcn_wmma_f32_16x16x32_bf16(
                false, ax, false, b_win, (short)0, cz, false, false);
#pragma unroll
        for (int kk = 0; kk < 4; ++kk)
            c = __builtin_amdgcn_wmma_f32_16x16x32_bf16(
                    false, ah[kk], false, b_wh[kk], (short)0, c, false, false);

        // -- hardtanh + write new h (C layout: VGPR r -> M=r | r+8) -----------
#pragma unroll
        for (int r = 0; r < 8; ++r) {
            float v = fminf(fmaxf(c[r], -1.0f), 1.0f);
            c[r] = v;
            hbuf[cur][r + (hi ? 8 : 0)][n0 + lrow] = (__bf16)v;
        }
        __syncthreads();   // one barrier/step thanks to double buffering

        p0 = q0; p1 = q1; p2 = q2;   // advance xs pipeline
    }

    // ---- output projection: out = h_T @ W_out^T (f32 accumulators) ----------
#pragma unroll
    for (int r = 0; r < 8; ++r)
        hf[r + (hi ? 8 : 0)][n0 + lrow] = c[r];
    __syncthreads();

    if (threadIdx.x < 16 * O_SZ) {
        const int m = threadIdx.x & 15;
        const int o = threadIdx.x >> 4;
        float acc = 0.0f;
#pragma unroll 4
        for (int k = 0; k < H_SZ; ++k)
            acc += hf[m][k] * W_out[o * H_SZ + k];
        out[(size_t)(b0 + m) * O_SZ + o] = acc;
    }
}

extern "C" void kernel_launch(void* const* d_in, const int* in_sizes, int n_in,
                              void* d_out, int out_size, void* d_ws, size_t ws_size,
                              hipStream_t stream) {
    (void)in_sizes; (void)n_in; (void)out_size; (void)d_ws; (void)ws_size;
    const float* xs    = (const float*)d_in[0];
    const float* W_in  = (const float*)d_in[1];
    const float* W_h   = (const float*)d_in[2];
    const float* W_out = (const float*)d_in[3];
    float* out = (float*)d_out;

    dim3 grid(B_SZ / 16);   // 64 independent batch tiles
    dim3 block(256);        // 8 wave32 waves
    rnn_hardtanh_fused<<<grid, block, 0, stream>>>(xs, W_in, W_h, W_out, out);
}
